// contrastive_kld_cosine_similarity_loss_31353261261534
// MI455X (gfx1250) — matrix operand order repaired and checked by reading, not compiled, not run on hardware
//
#include <hip/hip_runtime.h>
#include <hip/hip_bf16.h>

// ---------------------------------------------------------------------------
// contrastive KLD cosine-similarity loss, MI455X (gfx1250, wave32, WMMA)
//
// term1: KL(softmax(tn@tn^T) || softmax(pn@pn^T)) row-mean   (N=4096, D=1024)
// term2: KL(softmax(target)  || softmax(predicted)) row-mean
//
// Gram matrices are never materialized: cos-sim in [-1,1] lets us use a fixed
// shift (exp(s-1)) instead of online max, so each row only needs running sums
//   E1 = sum exp(s1-1), E2 = sum exp(s2-1), T = sum exp(s1-1)*(s1-s2)
//   KL_i = T/E1 + log(E2) - log(E1)
// B tiles are staged LDS via async DMA (global_load_async_to_lds_b128,
// ASYNCcnt) with double buffering; WMMAs consume from LDS (DScnt) while the
// next tile streams in. bf16 matrices (8MB each) are L2-resident (192MB L2).
// ---------------------------------------------------------------------------

#define N_ROWS 4096
#define D_DIM  1024
#define NCB    8      // column blocks (4096/512)
#define KSTEPS (D_DIM / 32)

typedef __attribute__((ext_vector_type(16))) __bf16 v16bf;
typedef __attribute__((ext_vector_type(8)))  float  v8f;

union FragBF { uint4 u[2]; v16bf v; };

__device__ __forceinline__ unsigned short f2bf(float f) {
    unsigned int u = __float_as_uint(f);
    u += 0x7FFFu + ((u >> 16) & 1u);           // round-to-nearest-even
    return (unsigned short)(u >> 16);
}

struct AddOp { __device__ float operator()(float a, float b) const { return a + b; } };
struct MaxOp { __device__ float operator()(float a, float b) const { return fmaxf(a, b); } };

template <class Op>
__device__ __forceinline__ float block_reduce256(float v, Op op) {
    __shared__ float sm[8];
    #pragma unroll
    for (int m = 16; m > 0; m >>= 1) v = op(v, __shfl_xor(v, m, 32));
    int w = threadIdx.x >> 5, l = threadIdx.x & 31;
    __syncthreads();
    if (l == 0) sm[w] = v;
    __syncthreads();
    v = sm[0];
    #pragma unroll
    for (int i = 1; i < 8; i++) v = op(v, sm[i]);
    return v;
}

// ---------------------------------------------------------------------------
// Kernel 1: per row -- L2 norm -> bf16 normalized copies, plus row KL over D
// ---------------------------------------------------------------------------
__global__ void prep_kernel(const float* __restrict__ target,
                            const float* __restrict__ predicted,
                            unsigned short* __restrict__ tn,
                            unsigned short* __restrict__ pn,
                            float* __restrict__ klrow) {
    const int row = blockIdx.x;
    const size_t base = (size_t)row * D_DIM;
    float t[4], p[4];
    #pragma unroll
    for (int i = 0; i < 4; i++) {
        int idx = threadIdx.x + i * 256;
        t[i] = target[base + idx];
        p[i] = predicted[base + idx];
    }
    float ssq_t = 0.f, ssq_p = 0.f, mx_t = -1e30f, mx_p = -1e30f;
    #pragma unroll
    for (int i = 0; i < 4; i++) {
        ssq_t += t[i] * t[i];  ssq_p += p[i] * p[i];
        mx_t = fmaxf(mx_t, t[i]); mx_p = fmaxf(mx_p, p[i]);
    }
    ssq_t = block_reduce256(ssq_t, AddOp());
    ssq_p = block_reduce256(ssq_p, AddOp());
    mx_t  = block_reduce256(mx_t,  MaxOp());
    mx_p  = block_reduce256(mx_p,  MaxOp());

    const float inv_t = 1.0f / fmaxf(__fsqrt_rn(ssq_t), 1e-8f);
    const float inv_p = 1.0f / fmaxf(__fsqrt_rn(ssq_p), 1e-8f);
    #pragma unroll
    for (int i = 0; i < 4; i++) {
        int idx = threadIdx.x + i * 256;
        tn[base + idx] = f2bf(t[i] * inv_t);
        pn[base + idx] = f2bf(p[i] * inv_p);
    }
    float et = 0.f, ep = 0.f;
    #pragma unroll
    for (int i = 0; i < 4; i++) { et += __expf(t[i] - mx_t); ep += __expf(p[i] - mx_p); }
    et = block_reduce256(et, AddOp());
    ep = block_reduce256(ep, AddOp());
    const float let = __logf(et), lep = __logf(ep);
    float kl = 0.f;
    #pragma unroll
    for (int i = 0; i < 4; i++) {
        float w     = __expf(t[i] - mx_t) / et;
        float log_t = t[i] - mx_t - let;
        float log_p = p[i] - mx_p - lep;
        kl += w * (log_t - log_p);
    }
    kl = block_reduce256(kl, AddOp());
    if (threadIdx.x == 0) klrow[row] = kl;
}

// ---------------------------------------------------------------------------
// Kernel 2: fused dual-Gram bf16 WMMA + softmax-KL partial sums.
// grid (32 row-blocks, 8 col-blocks), block 256 = 8 waves.
// Wave w: rows [bx*128 + w*16, +16). All waves share the 64-col B tile,
// staged in LDS by async DMA (double-buffered), consumed via ds_load_b128.
// ---------------------------------------------------------------------------
__global__ void __launch_bounds__(256)
gram_kl_kernel(const unsigned short* __restrict__ tn,
               const unsigned short* __restrict__ pn,
               float* __restrict__ E1,
               float* __restrict__ E2,
               float* __restrict__ Tm) {
    // double-buffered B tiles: [buf][matrix][64 cols][32 K] bf16 = 16 KB
    __shared__ __align__(16) unsigned short ldsB[2][2][64][32];

    const int tid  = threadIdx.x;
    const int wave = tid >> 5;
    const int lane = tid & 31;
    const int half = lane >> 4;       // 0: lanes 0-15, 1: lanes 16-31
    const int l16  = lane & 15;

    const int rowbase = blockIdx.x * 128 + wave * 16;
    const int cb      = blockIdx.y;
    const int colbase = cb * 512;

    // producer addressing (per thread): one b128 per matrix per K-step
    const int pcol = tid >> 2;          // 0..63
    const int pk   = (tid & 3) * 8;     // 0,8,16,24

    // A-fragment rows: lane l16 -> row rowbase+l16 (both halves, different K)
    const unsigned short* arow_t = tn + (size_t)(rowbase + l16) * D_DIM;
    const unsigned short* arow_p = pn + (size_t)(rowbase + l16) * D_DIM;

    float e1[8], e2[8], tt[8];
    #pragma unroll
    for (int r = 0; r < 8; r++) { e1[r] = 0.f; e2[r] = 0.f; tt[r] = 0.f; }

    for (int chunk = 0; chunk < 8; chunk++) {
        const int col0 = colbase + chunk * 64;
        const unsigned short* gbt = tn + (size_t)(col0 + pcol) * D_DIM + pk;
        const unsigned short* gbp = pn + (size_t)(col0 + pcol) * D_DIM + pk;

        __syncthreads();  // prior chunk's reads of buf0 complete before refill

        // prologue DMA: K-step 0 into buffer 0
        {
            unsigned dT = (unsigned)(uintptr_t)&ldsB[0][0][pcol][pk];
            unsigned dP = (unsigned)(uintptr_t)&ldsB[0][1][pcol][pk];
            unsigned long long sT = (unsigned long long)(uintptr_t)gbt;
            unsigned long long sP = (unsigned long long)(uintptr_t)gbp;
            asm volatile("global_load_async_to_lds_b128 %0, %1, off"
                         :: "v"(dT), "v"(sT) : "memory");
            asm volatile("global_load_async_to_lds_b128 %0, %1, off"
                         :: "v"(dP), "v"(sP) : "memory");
        }

        v8f c1[4], c2[4];
        #pragma unroll
        for (int tile = 0; tile < 4; tile++) { c1[tile] = (v8f)(0.f); c2[tile] = (v8f)(0.f); }

        for (int i = 0; i < KSTEPS; i++) {
            const int k0 = i * 32;
            asm volatile("s_wait_asynccnt 0x0" ::: "memory");  // buf[i&1] DMA done
            __syncthreads();                                   // visible to all waves

            if (i + 1 < KSTEPS) {  // kick DMA for next K-step into other buffer
                const int nb = (i + 1) & 1;
                unsigned dT = (unsigned)(uintptr_t)&ldsB[nb][0][pcol][pk];
                unsigned dP = (unsigned)(uintptr_t)&ldsB[nb][1][pcol][pk];
                unsigned long long sT = (unsigned long long)(uintptr_t)(gbt + k0 + 32);
                unsigned long long sP = (unsigned long long)(uintptr_t)(gbp + k0 + 32);
                asm volatile("global_load_async_to_lds_b128 %0, %1, off"
                             :: "v"(dT), "v"(sT) : "memory");
                asm volatile("global_load_async_to_lds_b128 %0, %1, off"
                             :: "v"(dP), "v"(sP) : "memory");
            }

            // A layout (16-bit 16x32): lanes 0-15 K=0..7 & 16..23,
            //                          lanes 16-31 K=8..15 & 24..31
            FragBF a1, a2;
            a1.u[0] = *(const uint4*)(arow_t + k0 + half * 8);
            a1.u[1] = *(const uint4*)(arow_t + k0 + 16 + half * 8);
            a2.u[0] = *(const uint4*)(arow_p + k0 + half * 8);
            a2.u[1] = *(const uint4*)(arow_p + k0 + 16 + half * 8);

            const int cur = i & 1;
            #pragma unroll
            for (int tile = 0; tile < 4; tile++) {
                // B layout (16-bit 32x16): lane n of col n, lanes 0-15 K=0..7 & 16..23, lanes 16-31 K=8..15 & 24..31
                const unsigned short* bt = &ldsB[cur][0][tile * 16 + l16][half * 8];
                const unsigned short* bp = &ldsB[cur][1][tile * 16 + l16][half * 8];
                FragBF b1, b2;
                b1.u[0] = *(const uint4*)(bt);
                b1.u[1] = *(const uint4*)(bt + 16);
                b2.u[0] = *(const uint4*)(bp);
                b2.u[1] = *(const uint4*)(bp + 16);
                c1[tile] = __builtin_amdgcn_wmma_f32_16x16x32_bf16(
                    false, a1.v, false, b1.v, (short)0, c1[tile], false, false);
                c2[tile] = __builtin_amdgcn_wmma_f32_16x16x32_bf16(
                    false, a2.v, false, b2.v, (short)0, c2[tile], false, false);
            }
        }

        // epilogue: fold 16x16 sim tiles into per-row running sums.
        // C layout: VGPR r, lanes 0-15 -> (M=r, N=lane); lanes 16-31 -> (M=8+r, N=lane-16)
        #pragma unroll
        for (int tile = 0; tile < 4; tile++) {
            #pragma unroll
            for (int r = 0; r < 8; r++) {
                float s1 = c1[tile][r];
                float s2 = c2[tile][r];
                float w1 = __expf(s1 - 1.0f);
                e1[r] += w1;
                e2[r] += __expf(s2 - 1.0f);
                tt[r] += w1 * (s1 - s2);
            }
        }
    }

    // reduce across the 16 lanes of each half (each half covers all 16 N's)
    #pragma unroll
    for (int r = 0; r < 8; r++) {
        #pragma unroll
        for (int m = 8; m > 0; m >>= 1) {
            e1[r] += __shfl_xor(e1[r], m, 32);
            e2[r] += __shfl_xor(e2[r], m, 32);
            tt[r] += __shfl_xor(tt[r], m, 32);
        }
    }
    if (l16 == 0) {
        // half 0 -> rows rowbase+0..7 ; half 1 -> rows rowbase+8..15
        #pragma unroll
        for (int r = 0; r < 8; r++) {
            int row = rowbase + half * 8 + r;
            int idx = cb * N_ROWS + row;
            E1[idx] = e1[r];
            E2[idx] = e2[r];
            Tm[idx] = tt[r];
        }
    }
}

// ---------------------------------------------------------------------------
// Kernel 3: deterministic finalize.
// ---------------------------------------------------------------------------
__global__ void finalize_kernel(const float* __restrict__ E1,
                                const float* __restrict__ E2,
                                const float* __restrict__ Tm,
                                const float* __restrict__ klrow,
                                float* __restrict__ out) {
    float acc = 0.f;
    for (int row = threadIdx.x; row < N_ROWS; row += 256) {
        float e1 = 0.f, e2 = 0.f, t = 0.f;
        #pragma unroll
        for (int c = 0; c < NCB; c++) {
            e1 += E1[c * N_ROWS + row];
            e2 += E2[c * N_ROWS + row];
            t  += Tm[c * N_ROWS + row];
        }
        acc += t / e1 + __logf(e2) - __logf(e1);
        acc += klrow[row];
    }
    acc = block_reduce256(acc, AddOp());
    if (threadIdx.x == 0) out[0] = acc / (float)N_ROWS;
}

extern "C" void kernel_launch(void* const* d_in, const int* in_sizes, int n_in,
                              void* d_out, int out_size, void* d_ws, size_t ws_size,
                              hipStream_t stream) {
    (void)in_sizes; (void)n_in; (void)out_size; (void)ws_size;
    const float* target    = (const float*)d_in[0];
    const float* predicted = (const float*)d_in[1];
    float* out = (float*)d_out;

    // workspace layout: [tn bf16 8MB][pn bf16 8MB][E1][E2][T][klrow]
    unsigned short* tn = (unsigned short*)d_ws;
    unsigned short* pn = tn + (size_t)N_ROWS * D_DIM;
    float* fb    = (float*)(pn + (size_t)N_ROWS * D_DIM);
    float* E1    = fb;
    float* E2    = E1 + NCB * N_ROWS;
    float* Tm    = E2 + NCB * N_ROWS;
    float* klrow = Tm + NCB * N_ROWS;

    prep_kernel<<<N_ROWS, 256, 0, stream>>>(target, predicted, tn, pn, klrow);
    gram_kl_kernel<<<dim3(32, NCB), 256, 0, stream>>>(tn, pn, E1, E2, Tm);
    finalize_kernel<<<1, 256, 0, stream>>>(E1, E2, Tm, klrow, out);
}
